// ROAD_28492813042070
// MI455X (gfx1250) — compile-verified, matrix operand order checked by reading, not run on hardware
//
#include <hip/hip_runtime.h>
#include <stdint.h>

// ---------------------------------------------------------------------------
// Problem constants (from reference)
// ---------------------------------------------------------------------------
#define BATCH   16
#define CHANS   3
#define HH      512
#define WW      512
#define HW      (HH * WW)                 // 262144
#define N_IMG   (BATCH * CHANS * HW)      // 12,582,912
#define N_MASK  (BATCH * HW)              // 4,194,304
#define K_SEL   78643                     // int(round(0.3*512*512))
#define NOISE_L 0.01f
#define TOL2    1e-14f                    // (1e-7)^2, relative to ||b||^2
#define MAXIT   300

#define W_E (1.0f / 6.0f)
#define W_C (1.0f / 12.0f)

// ---------------------------------------------------------------------------
// Workspace layout (bytes). Total required ~151.3 MB.
// ---------------------------------------------------------------------------
#define OFF_SCAL    0u                         // struct Scal (zeroed)
#define OFF_HISTC   256u                       // u32[16][1024]
#define OFF_HISTF   (OFF_HISTC + 65536u)       // u32[16][1024]
#define OFF_BSTAR   (OFF_HISTF + 65536u)       // i32[16]
#define OFF_REMAIN  (OFF_BSTAR + 64u)          // i32[16]
#define OFF_THRT    (OFF_REMAIN + 64u)         // i32[16]
#define OFF_QUOTA   (OFF_THRT + 64u)           // u32[16]
#define ZERO_BYTES  (OFF_QUOTA + 64u)          // memset [0, ZERO_BYTES)
#define OFF_TIECNT  ZERO_BYTES                 // u32[16][256]
#define OFF_TIEBASE (OFF_TIECNT + 16384u)      // u32[16][256]
#define OFF_PART    (OFF_TIEBASE + 16384u)     // float[16384]
#define OFF_R       (OFF_PART + 65536u)        // float[N_IMG]  (256B aligned)
#define OFF_P       (OFF_R + (size_t)N_IMG * 4u)
#define OFF_AP      (OFF_P + (size_t)N_IMG * 4u)

struct Scal {
    unsigned done;
    float    rr_old;
    float    rr_new;
    float    pAp;
    float    bnorm2;
};

typedef __attribute__((ext_vector_type(16))) _Float16 v16h;
typedef __attribute__((ext_vector_type(8)))  float    v8f;

// ---------------------------------------------------------------------------
// CDNA5 feature probes
// ---------------------------------------------------------------------------
#if __has_builtin(__builtin_amdgcn_wmma_f32_16x16x32_f16)
#define USE_WMMA_RED 1
#pragma message("CDNA5: WMMA f32_16x16x32_f16 reduction path ENABLED")
#else
#pragma message("CDNA5: WMMA builtin NOT found -> shuffle reduction fallback")
#endif

#if __has_builtin(__builtin_amdgcn_global_load_async_to_lds_b32) && \
    __has_builtin(__builtin_amdgcn_s_wait_asynccnt)
#define USE_ASYNC_LDS 1
#pragma message("CDNA5: global_load_async_to_lds_b32 path ENABLED")
#else
#pragma message("CDNA5: async-to-LDS builtins NOT found -> plain LDS staging")
#endif

#if __has_builtin(__builtin_amdgcn_global_load_async_to_lds_b128) && \
    __has_builtin(__builtin_amdgcn_s_wait_asynccnt)
#define USE_ASYNC_B128 1
#pragma message("CDNA5: global_load_async_to_lds_b128 (vectorized) path ENABLED")
#endif

#if defined(USE_ASYNC_LDS) || defined(USE_ASYNC_B128)
typedef __attribute__((address_space(1))) int g_int_t;
typedef __attribute__((address_space(3))) int l_int_t;
#endif

#ifdef USE_ASYNC_LDS
__device__ __forceinline__ void async_ld_f32(float* lds_dst, const float* gsrc) {
    // generic->AS casts via integer (LDS flat addr low 32 bits == LDS offset)
    l_int_t* ldst = (l_int_t*)(unsigned)(uintptr_t)lds_dst;
    g_int_t* gptr = (g_int_t*)(uintptr_t)gsrc;
    __builtin_amdgcn_global_load_async_to_lds_b32(gptr, ldst, 0, 0);
}
#endif

#ifdef USE_ASYNC_B128
// b128 variant wants an int4-vector pointee (pointee width selects transfer width)
typedef int v4i_vec __attribute__((vector_size(16)));
typedef __attribute__((address_space(1))) v4i_vec g_v4i_t;
typedef __attribute__((address_space(3))) v4i_vec l_v4i_t;
__device__ __forceinline__ void async_ld_f32x4(float* lds_dst, const float* gsrc) {
    l_v4i_t* ldst = (l_v4i_t*)(unsigned)(uintptr_t)lds_dst;
    g_v4i_t* gptr = (g_v4i_t*)(uintptr_t)gsrc;
    __builtin_amdgcn_global_load_async_to_lds_b128(gptr, ldst, 0, 0);
}
#endif

// Non-temporal hints for single-use streams (keep r/p/Ap/x hot in the 192MB L2)
#if __has_builtin(__builtin_nontemporal_load)
#define NT_LOAD(p)     __builtin_nontemporal_load(p)
#define NT_STORE(p, v) __builtin_nontemporal_store((v), (p))
#else
#define NT_LOAD(p)     (*(p))
#define NT_STORE(p, v) (*(p) = (v))
#endif

// ---------------------------------------------------------------------------
// Deterministic 256-thread block sum on the matrix pipe: 256 f32 partials are
// split hi/lo into two f16 A tiles (each fills half of a 16x32 A), multiplied by
// an all-ones B, accumulated in f32 C. With B == 1 the (m,k) placement is
// irrelevant: D[m][n] = rowsum(m). 8 register adds + one cross-half shuffle
// finish the sum. ~22-bit effective input mantissa, exact f32 accumulation.
// ---------------------------------------------------------------------------
__device__ __forceinline__ float block_sum_256(float v, float* s_red) {
    const int tid = (int)(threadIdx.x + threadIdx.y * blockDim.x);
    s_red[tid] = v;
    __syncthreads();
    float total = 0.0f;
    if (tid < 32) {
#ifdef USE_WMMA_RED
        v16h ahi, alo, ones;
#pragma unroll
        for (int j = 0; j < 16; ++j) {
            ahi[j] = (_Float16)0.0f; alo[j] = (_Float16)0.0f; ones[j] = (_Float16)1.0f;
        }
#pragma unroll
        for (int j = 0; j < 8; ++j) {
            float x = s_red[tid * 8 + j];
            _Float16 h = (_Float16)x;
            ahi[j] = h;
            alo[j] = (_Float16)(x - (float)h);
        }
        v8f c = {0.f, 0.f, 0.f, 0.f, 0.f, 0.f, 0.f, 0.f};
        c = __builtin_amdgcn_wmma_f32_16x16x32_f16(false, ahi, false, ones, (short)0, c, false, false);
        c = __builtin_amdgcn_wmma_f32_16x16x32_f16(false, alo, false, ones, (short)0, c, false, false);
        float acc = c[0] + c[1] + c[2] + c[3] + c[4] + c[5] + c[6] + c[7];
        acc += __shfl_xor(acc, 16, 32);
        total = acc;
#else
        float acc = 0.0f;
#pragma unroll
        for (int j = 0; j < 8; ++j) acc += s_red[tid * 8 + j];
#pragma unroll
        for (int off = 16; off > 0; off >>= 1) acc += __shfl_down(acc, off, 32);
        acc = __shfl(acc, 0, 32);
        total = acc;
#endif
    }
    __syncthreads();
    return total;   // valid at tid == 0
}

// ---------------------------------------------------------------------------
// Top-k threshold selection (two-level counting select, exact count,
// index-ordered tie break == stable argsort of the reference)
// ---------------------------------------------------------------------------
__global__ __launch_bounds__(256) void k_hist_coarse(const int* __restrict__ smap,
                                                     unsigned* __restrict__ histC) {
    __shared__ unsigned h[1024];
    const int b = blockIdx.y, blk = blockIdx.x, tid = threadIdx.x;
    for (int i = tid; i < 1024; i += 256) h[i] = 0u;
    __syncthreads();
    const int4 v = ((const int4*)(smap + (size_t)b * HW))[(size_t)blk * 256 + tid];
    atomicAdd(&h[((unsigned)v.x) >> 10], 1u);
    atomicAdd(&h[((unsigned)v.y) >> 10], 1u);
    atomicAdd(&h[((unsigned)v.z) >> 10], 1u);
    atomicAdd(&h[((unsigned)v.w) >> 10], 1u);
    __syncthreads();
    unsigned* gh = histC + (size_t)b * 1024;
    for (int i = tid; i < 1024; i += 256)
        if (h[i]) atomicAdd(&gh[i], h[i]);
}

__global__ void k_resolve_coarse(const unsigned* __restrict__ histC,
                                 int* __restrict__ bstar, int* __restrict__ remain) {
    const int b = blockIdx.x;
    if (threadIdx.x != 0) return;
    const unsigned* h = histC + (size_t)b * 1024;
    unsigned cum = 0;
    int bs = 0, rem = 0;
    for (int i = 1023; i >= 0; --i) {
        unsigned c = h[i];
        if (cum + c >= (unsigned)K_SEL) { bs = i; rem = (int)((unsigned)K_SEL - cum); break; }
        cum += c;
    }
    bstar[b] = bs; remain[b] = rem;
}

__global__ __launch_bounds__(256) void k_hist_fine(const int* __restrict__ smap,
                                                   const int* __restrict__ bstar,
                                                   unsigned* __restrict__ histF) {
    __shared__ unsigned h[1024];
    const int b = blockIdx.y, blk = blockIdx.x, tid = threadIdx.x;
    const int bs = bstar[b];
    for (int i = tid; i < 1024; i += 256) h[i] = 0u;
    __syncthreads();
    const int4 v = ((const int4*)(smap + (size_t)b * HW))[(size_t)blk * 256 + tid];
    int vv[4] = {v.x, v.y, v.z, v.w};
#pragma unroll
    for (int j = 0; j < 4; ++j)
        if ((vv[j] >> 10) == bs) atomicAdd(&h[vv[j] & 1023], 1u);
    __syncthreads();
    unsigned* gh = histF + (size_t)b * 1024;
    for (int i = tid; i < 1024; i += 256)
        if (h[i]) atomicAdd(&gh[i], h[i]);
}

__global__ void k_resolve_fine(const unsigned* __restrict__ histF,
                               const int* __restrict__ bstar, const int* __restrict__ remain,
                               int* __restrict__ thrT, unsigned* __restrict__ quota) {
    const int b = blockIdx.x;
    if (threadIdx.x != 0) return;
    const unsigned* h = histF + (size_t)b * 1024;
    const unsigned rem = (unsigned)remain[b];
    unsigned cum = 0; int l = 0; unsigned q = 0;
    for (int i = 1023; i >= 0; --i) {
        unsigned c = h[i];
        if (cum + c >= rem) { l = i; q = rem - cum; break; }
        cum += c;
    }
    thrT[b] = (bstar[b] << 10) | l;
    quota[b] = q;
}

__global__ __launch_bounds__(256) void k_tie_count(const int* __restrict__ smap,
                                                   const int* __restrict__ thrT,
                                                   unsigned* __restrict__ tieCnt) {
    __shared__ unsigned ss[256];
    const int b = blockIdx.y, blk = blockIdx.x, tid = threadIdx.x;
    const int T = thrT[b];
    const int4 v = ((const int4*)(smap + (size_t)b * HW))[(size_t)blk * 256 + tid];
    unsigned ct = (unsigned)(v.x == T) + (unsigned)(v.y == T) +
                  (unsigned)(v.z == T) + (unsigned)(v.w == T);
    ss[tid] = ct;
    __syncthreads();
    for (int off = 128; off > 0; off >>= 1) {
        if (tid < off) ss[tid] += ss[tid + off];
        __syncthreads();
    }
    if (tid == 0) tieCnt[(size_t)b * 256 + blk] = ss[0];
}

__global__ void k_tie_scan(const unsigned* __restrict__ tieCnt, unsigned* __restrict__ tieBase) {
    const int b = blockIdx.x;
    if (threadIdx.x != 0) return;
    unsigned cum = 0;
    for (int i = 0; i < 256; ++i) {
        tieBase[(size_t)b * 256 + i] = cum;
        cum += tieCnt[(size_t)b * 256 + i];
    }
}

__global__ __launch_bounds__(256) void k_mask_build(const int* __restrict__ smap,
                                                    const int* __restrict__ thrT,
                                                    const unsigned* __restrict__ quota,
                                                    const unsigned* __restrict__ tieBase,
                                                    float* __restrict__ mask) {
    __shared__ unsigned ss[256];
    const int b = blockIdx.y, blk = blockIdx.x, tid = threadIdx.x;
    const int T = thrT[b];
    const unsigned q = quota[b];
    const int4 v = ((const int4*)(smap + (size_t)b * HW))[(size_t)blk * 256 + tid];
    int vv[4] = {v.x, v.y, v.z, v.w};
    unsigned ct = 0;
#pragma unroll
    for (int j = 0; j < 4; ++j) ct += (unsigned)(vv[j] == T);
    // index-ordered exclusive scan of tie counts across the block
    ss[tid] = ct;
    __syncthreads();
    for (int off = 1; off < 256; off <<= 1) {
        unsigned t = (tid >= off) ? ss[tid - off] : 0u;
        __syncthreads();
        ss[tid] += t;
        __syncthreads();
    }
    unsigned rank = tieBase[(size_t)b * 256 + blk] + (ss[tid] - ct);
    float mv[4];
#pragma unroll
    for (int j = 0; j < 4; ++j) {
        float m = 1.0f;
        if (vv[j] > T) m = 0.0f;
        else if (vv[j] == T) { if (rank < q) m = 0.0f; rank++; }
        mv[j] = m;
    }
    float4 o; o.x = mv[0]; o.y = mv[1]; o.z = mv[2]; o.w = mv[3];
    ((float4*)(mask + (size_t)b * HW))[(size_t)blk * 256 + tid] = o;
}

// ---------------------------------------------------------------------------
// Fused 9-point stencil. mode 0: r = p = unk*stencil(m*img), partial = sum r^2
//                        mode 1: Ap = unk>0 ? S*p - stencil(unk*p) : p,
//                                partial = sum p*Ap
// 32x32 tile + halo staged in LDS. The LDS window is 40 columns wide and
// 16B-aligned (cols bx*32-4 .. bx*32+35); since 512 and 32 are multiples of 4,
// every 16B segment is fully in-image or fully out, enabling pure b128 async
// transfers (340 segments/array/tile vs 1156 b32 gathers).
// ---------------------------------------------------------------------------
__global__ __launch_bounds__(256) void k_stencil(int mode,
                                                 const float* __restrict__ field,
                                                 const float* __restrict__ mask,
                                                 float* __restrict__ out,
                                                 float* __restrict__ out2,
                                                 float* __restrict__ partials,
                                                 const Scal* __restrict__ sc) {
    if (mode == 1 && sc->done) return;
    __shared__ __attribute__((aligned(16))) float ls_f[34][40];
    __shared__ __attribute__((aligned(16))) float ls_m[34][40];
    __shared__ float s_red[256];
    const int bx = blockIdx.x, by = blockIdx.y, z = blockIdx.z;
    const int b = z / CHANS;
    const int tx = threadIdx.x, ty = threadIdx.y;
    const int tid = ty * 32 + tx;
    const float* fptr = field + (size_t)z * HW;
    const float* mptr = mask + (size_t)b * HW;
    const int oy = by * 32 - 1;          // first halo row
    const int obase = bx * 32 - 4;       // aligned first column of LDS window

#ifdef USE_ASYNC_B128
    for (int s = tid; s < 34 * 10; s += 256) {
        const int row = s / 10, c4 = s - row * 10;
        const int gy = oy + row;
        const int gc = obase + c4 * 4;
        const bool inb = ((unsigned)gy < (unsigned)HH) && ((unsigned)gc <= (unsigned)(WW - 4));
        float* df = &ls_f[row][c4 * 4];
        float* dm = &ls_m[row][c4 * 4];
        if (inb) {
            const size_t g = (size_t)gy * WW + gc;
            async_ld_f32x4(df, fptr + g);
            async_ld_f32x4(dm, mptr + g);
        } else {
            float4 zz; zz.x = 0.f; zz.y = 0.f; zz.z = 0.f; zz.w = 0.f;
            *(float4*)df = zz;
            *(float4*)dm = zz;
        }
    }
    __builtin_amdgcn_s_wait_asynccnt(0);
#else
    // scalar staging of the 34x34 region actually referenced (cols 3..36)
    for (int i = tid; i < 34 * 34; i += 256) {
        const int row = i / 34, c = (i - row * 34) + 3;
        const int gy = oy + row, gx = obase + c;
        const bool inb = ((unsigned)gy < (unsigned)HH) && ((unsigned)gx < (unsigned)WW);
        if (inb) {
#ifdef USE_ASYNC_LDS
            async_ld_f32(&ls_f[row][c], fptr + (size_t)gy * WW + gx);
            async_ld_f32(&ls_m[row][c], mptr + (size_t)gy * WW + gx);
#else
            ls_f[row][c] = fptr[(size_t)gy * WW + gx];
            ls_m[row][c] = mptr[(size_t)gy * WW + gx];
#endif
        } else {
            ls_f[row][c] = 0.0f;
            ls_m[row][c] = 0.0f;
        }
    }
#ifdef USE_ASYNC_LDS
    __builtin_amdgcn_s_wait_asynccnt(0);
#endif
#endif
    __syncthreads();

    float partial = 0.0f;
#pragma unroll
    for (int rr = 0; rr < 4; ++rr) {
        const int y = ty + rr * 8, x = tx;
        const int ly = y + 1, lx = x + 4;     // center at col x+4 in 40-wide window
        const int gy = by * 32 + y, gx = bx * 32 + x;
        const float cf = ls_f[ly][lx];
        const float cm = ls_m[ly][lx];
        float sten = 0.0f, S = 0.0f;
#define NB(dy, dx, w)                                                            \
        {                                                                        \
            float nf = ls_f[ly + (dy)][lx + (dx)];                               \
            float nm = ls_m[ly + (dy)][lx + (dx)];                               \
            float g  = (mode == 0) ? (nm * nf) : ((1.0f - nm) * nf);             \
            sten += (w) * g;                                                     \
            if (((unsigned)(gy + (dy)) < (unsigned)HH) &&                        \
                ((unsigned)(gx + (dx)) < (unsigned)WW)) S += (w);                \
        }
        NB(-1, -1, W_C) NB(-1, 0, W_E) NB(-1, 1, W_C)
        NB(0, -1, W_E)                 NB(0, 1, W_E)
        NB(1, -1, W_C)  NB(1, 0, W_E)  NB(1, 1, W_C)
#undef NB
        const size_t idx = (size_t)z * HW + (size_t)gy * WW + gx;
        if (mode == 0) {
            const float val = (1.0f - cm) * sten;
            out[idx] = val;
            out2[idx] = val;
            partial += val * val;
        } else {
            const float unkc = 1.0f - cm;
            const float val = (unkc > 0.0f) ? (S * cf - sten) : cf;
            out[idx] = val;
            partial += cf * val;
        }
    }
    const float tot = block_sum_256(partial, s_red);
    if (tid == 0)
        partials[(size_t)(z * gridDim.y + by) * gridDim.x + bx] = tot;
}

// ---------------------------------------------------------------------------
// CG support kernels
// ---------------------------------------------------------------------------
__global__ __launch_bounds__(256) void k_reduce(const float* __restrict__ in, int n,
                                                float* __restrict__ outp,
                                                const Scal* __restrict__ sc) {
    if (sc->done) return;
    __shared__ float s_red[256];
    const int tid = threadIdx.x;
    float local = 0.0f;
    for (int i = tid; i < n; i += 256) local += in[i];
    const float tot = block_sum_256(local, s_red);
    if (tid == 0) *outp = tot;
}

__global__ void k_init_b(Scal* sc) {
    sc->bnorm2 = sc->rr_new;
    sc->rr_old = sc->rr_new;
}

__global__ void k_cg_pre(Scal* sc) {
    if (sc->done) return;
    if (sc->rr_new <= TOL2 * sc->bnorm2) { sc->done = 1u; return; }
    sc->rr_old = sc->rr_new;
}

__global__ __launch_bounds__(256) void k_update1(float* __restrict__ x, float* __restrict__ r,
                                                 const float* __restrict__ p,
                                                 const float* __restrict__ Ap,
                                                 const Scal* __restrict__ sc,
                                                 float* __restrict__ partials) {
    if (sc->done) return;
    __shared__ float s_red[256];
    const float alpha = sc->rr_old / sc->pAp;
    const int tid = threadIdx.x;
    const size_t base = (size_t)blockIdx.x * 2048 + tid;
    float partial = 0.0f;
#pragma unroll
    for (int j = 0; j < 8; ++j) {
        const size_t i = base + (size_t)j * 256;
        const float pi = p[i];
        const float api = NT_LOAD(Ap + i);      // Ap is dead after this kernel
        x[i] += alpha * pi;
        const float ri = r[i] - alpha * api;
        r[i] = ri;
        partial += ri * ri;
    }
    const float tot = block_sum_256(partial, s_red);
    if (tid == 0) partials[blockIdx.x] = tot;
}

__global__ __launch_bounds__(256) void k_pupdate(float* __restrict__ p,
                                                 const float* __restrict__ r,
                                                 const Scal* __restrict__ sc) {
    if (sc->done) return;
    const float beta = sc->rr_new / sc->rr_old;
    const int tid = threadIdx.x;
    const size_t base = (size_t)blockIdx.x * 2048 + tid;
#pragma unroll
    for (int j = 0; j < 8; ++j) {
        const size_t i = base + (size_t)j * 256;
        __builtin_prefetch(p + i + 2048, 0, 0);   // -> global_prefetch_b8
        p[i] = r[i] + beta * p[i];
    }
}

__global__ __launch_bounds__(256) void k_compose(float* __restrict__ io,
                                                 const float* __restrict__ img,
                                                 const float* __restrict__ mask,
                                                 const float* __restrict__ noise) {
    const int tid = threadIdx.x;
    const size_t base = (size_t)blockIdx.x * 2048 + tid;
#pragma unroll
    for (int j = 0; j < 8; ++j) {
        const size_t i = base + (size_t)j * 256;
        const size_t z = i / HW;          // (b*C + c)
        const size_t b = z / CHANS;
        const size_t pix = i - z * HW;
        const float m  = NT_LOAD(mask + b * HW + pix);
        const float im = NT_LOAD(img + i);
        const float nz = NT_LOAD(noise + i);
        const float xi = io[i];
        NT_STORE(io + i, m * im + (1.0f - m) * (xi + NOISE_L * nz));
    }
}

// ---------------------------------------------------------------------------
// Host-side orchestration (graph-capture safe: only kernel launches + async memset)
// ---------------------------------------------------------------------------
extern "C" void kernel_launch(void* const* d_in, const int* in_sizes, int n_in,
                              void* d_out, int out_size, void* d_ws, size_t ws_size,
                              hipStream_t stream) {
    (void)in_sizes; (void)n_in; (void)out_size; (void)ws_size;

    const float* img   = (const float*)d_in[0];
    const int*   smap  = (const int*)d_in[1];
    const float* noise = (const float*)d_in[2];

    float* x_out    = (float*)d_out;              // imputed (also CG solution x)
    float* mask_out = (float*)d_out + N_IMG;      // mask output

    unsigned char* ws = (unsigned char*)d_ws;
    Scal*     sc      = (Scal*)(ws + OFF_SCAL);
    unsigned* histC   = (unsigned*)(ws + OFF_HISTC);
    unsigned* histF   = (unsigned*)(ws + OFF_HISTF);
    int*      bstar   = (int*)(ws + OFF_BSTAR);
    int*      remain  = (int*)(ws + OFF_REMAIN);
    int*      thrT    = (int*)(ws + OFF_THRT);
    unsigned* quota   = (unsigned*)(ws + OFF_QUOTA);
    unsigned* tieCnt  = (unsigned*)(ws + OFF_TIECNT);
    unsigned* tieBase = (unsigned*)(ws + OFF_TIEBASE);
    float*    part    = (float*)(ws + OFF_PART);
    float*    r_f     = (float*)(ws + OFF_R);
    float*    p_f     = (float*)(ws + OFF_P);
    float*    ap_f    = (float*)(ws + OFF_AP);

    // Zero control block + histograms; zero x (solution lives in d_out).
    (void)hipMemsetAsync(ws, 0, ZERO_BYTES, stream);
    (void)hipMemsetAsync(d_out, 0, (size_t)N_IMG * sizeof(float), stream);

    const dim3 gSel(256, BATCH), bSel(256);
    const dim3 gSten(WW / 32, HH / 32, BATCH * CHANS), bSten(32, 8);
    const int nStenBlocks = (WW / 32) * (HH / 32) * BATCH * CHANS;   // 12288
    const int nVecBlocks  = N_IMG / 2048;                            // 6144

    // ---- top-k mask (stable index-ordered tie break) ----
    k_hist_coarse<<<gSel, bSel, 0, stream>>>(smap, histC);
    k_resolve_coarse<<<BATCH, 32, 0, stream>>>(histC, bstar, remain);
    k_hist_fine<<<gSel, bSel, 0, stream>>>(smap, bstar, histF);
    k_resolve_fine<<<BATCH, 32, 0, stream>>>(histF, bstar, remain, thrT, quota);
    k_tie_count<<<gSel, bSel, 0, stream>>>(smap, thrT, tieCnt);
    k_tie_scan<<<BATCH, 32, 0, stream>>>(tieCnt, tieBase);
    k_mask_build<<<gSel, bSel, 0, stream>>>(smap, thrT, quota, tieBase, mask_out);

    // ---- CG init: r = p = rhs = unk*stencil(m*img); bnorm2 = ||r||^2 ----
    k_stencil<<<gSten, bSten, 0, stream>>>(0, img, mask_out, r_f, p_f, part, sc);
    k_reduce<<<1, 256, 0, stream>>>(part, nStenBlocks, &sc->rr_new, sc);
    k_init_b<<<1, 1, 0, stream>>>(sc);

    // ---- CG loop (fixed launches, device-side convergence early-exit) ----
    for (int it = 0; it < MAXIT; ++it) {
        k_cg_pre<<<1, 1, 0, stream>>>(sc);
        k_stencil<<<gSten, bSten, 0, stream>>>(1, p_f, mask_out, ap_f, nullptr, part, sc);
        k_reduce<<<1, 256, 0, stream>>>(part, nStenBlocks, &sc->pAp, sc);
        k_update1<<<nVecBlocks, 256, 0, stream>>>(x_out, r_f, p_f, ap_f, sc, part);
        k_reduce<<<1, 256, 0, stream>>>(part, nVecBlocks, &sc->rr_new, sc);
        k_pupdate<<<nVecBlocks, 256, 0, stream>>>(p_f, r_f, sc);
    }

    // ---- out = m*img + unk*(x + 0.01*noise) (in place over x) ----
    k_compose<<<nVecBlocks, 256, 0, stream>>>(x_out, img, mask_out, noise);
}
